// SpaceTimeCLAM_TSSM_28389733827325
// MI455X (gfx1250) — matrix operand order repaired and checked by reading, not compiled, not run on hardware
//
#include <hip/hip_runtime.h>
#include <hip/hip_bf16.h>

// ---------------- CDNA5 WMMA types ----------------
typedef __attribute__((ext_vector_type(16))) _Float16 v16h;
typedef __attribute__((ext_vector_type(8)))  float    v8f;

#define WMMA_F16(a,b,c) __builtin_amdgcn_wmma_f32_16x16x32_f16(false,(a),false,(b),(short)0,(c),false,false)

// model dims
#define BB 16
#define TT 50
#define AA 6
#define SSq 49            // S = T-1
#define DM 512
#define NH 8
#define DH 64
#define EMB 1024
#define SSD 1024          // SS*SD
#define DENSE_IN 1536

__device__ __forceinline__ float siluf(float x) { return x / (1.0f + __expf(-x)); }

// Load one lane's A fragment (f16 16x32 layout): two contiguous 8-float blocks
// base must already include +8*half; reads floats [0..7] and [16..23].
__device__ __forceinline__ v16h load_afrag(const float* __restrict__ base)
{
    const float4* p = (const float4*)base;
    float4 a = p[0], b = p[1], c = p[4], d = p[5];
    v16h f;
    f[0]  = (_Float16)a.x; f[1]  = (_Float16)a.y; f[2]  = (_Float16)a.z; f[3]  = (_Float16)a.w;
    f[4]  = (_Float16)b.x; f[5]  = (_Float16)b.y; f[6]  = (_Float16)b.z; f[7]  = (_Float16)b.w;
    f[8]  = (_Float16)c.x; f[9]  = (_Float16)c.y; f[10] = (_Float16)c.z; f[11] = (_Float16)c.w;
    f[12] = (_Float16)d.x; f[13] = (_Float16)d.y; f[14] = (_Float16)d.z; f[15] = (_Float16)d.w;
    return f;
}

// =====================================================================
// Generic WMMA GEMM: C[M,N] = act( A[M,Kpad](f32) * B[Kpad,Npad](f16) + bias + res )
// one wave computes a 32(M) x 32(N) tile: 4 accumulators, 4 WMMA per K-step.
// gridDim.x = ceil(M/32), gridDim.y = ceil(N/32)
// =====================================================================
__global__ __launch_bounds__(32) void k_gemm(
    const float* __restrict__ Am, const _Float16* __restrict__ Bm,
    const float* __restrict__ bias, const float* __restrict__ res,
    float* __restrict__ Cm, int M, int N, int Npad, int Kpad, int act)
{
    int m0 = blockIdx.x * 32;
    int n0 = blockIdx.y * 32;
    int lane = threadIdx.x;
    int half = lane >> 4;
    int l16  = lane & 15;

    int ar0 = m0 + l16;       int ar0c = (ar0 < M) ? ar0 : (M - 1);
    int ar1 = m0 + 16 + l16;  int ar1c = (ar1 < M) ? ar1 : (M - 1);
    const float*    a0p = Am + (size_t)ar0c * Kpad + 8 * half;
    const float*    a1p = Am + (size_t)ar1c * Kpad + 8 * half;
    const _Float16* bp  = Bm + (size_t)(8 * half) * Npad + n0 + l16;
    const size_t    bstep = (size_t)32 * Npad;

    v8f acc00 = {}, acc01 = {}, acc10 = {}, acc11 = {};
    for (int k0 = 0; k0 < Kpad; k0 += 32) {
        v16h af0 = load_afrag(a0p);
        v16h af1 = load_afrag(a1p);
        v16h bf0, bf1;
#pragma unroll
        for (int s = 0; s < 16; ++s) { const _Float16* q = bp + (size_t)((s < 8) ? s : s + 8) * Npad; bf0[s] = q[0]; bf1[s] = q[16]; }
        __builtin_prefetch((const void*)(bp + bstep), 0, 1);
        acc00 = WMMA_F16(af0, bf0, acc00);
        acc01 = WMMA_F16(af0, bf1, acc01);
        acc10 = WMMA_F16(af1, bf0, acc10);
        acc11 = WMMA_F16(af1, bf1, acc11);
        a0p += 32; a1p += 32; bp += bstep;
    }

    int c0 = n0 + l16, c1 = n0 + 16 + l16;
    float bs0 = (bias && c0 < N) ? bias[c0] : 0.0f;
    float bs1 = (bias && c1 < N) ? bias[c1] : 0.0f;
#pragma unroll
    for (int j = 0; j < 8; ++j) {
        int r0 = m0 + j + 8 * half;
        int r1 = r0 + 16;
        if (r0 < M) {
            if (c0 < N) {
                float x = acc00[j] + bs0;
                if (res) x += res[(size_t)r0 * N + c0];
                if (act) x = siluf(x);
                Cm[(size_t)r0 * N + c0] = x;
            }
            if (c1 < N) {
                float x = acc01[j] + bs1;
                if (res) x += res[(size_t)r0 * N + c1];
                if (act) x = siluf(x);
                Cm[(size_t)r0 * N + c1] = x;
            }
        }
        if (r1 < M) {
            if (c0 < N) {
                float x = acc10[j] + bs0;
                if (res) x += res[(size_t)r1 * N + c0];
                if (act) x = siluf(x);
                Cm[(size_t)r1 * N + c0] = x;
            }
            if (c1 < N) {
                float x = acc11[j] + bs1;
                if (res) x += res[(size_t)r1 * N + c1];
                if (act) x = siluf(x);
                Cm[(size_t)r1 * N + c1] = x;
            }
        }
    }
}

// =====================================================================
// Implicit-GEMM conv / deconv (4x4, stride 2, SAME), NHWC.
// A gathered on the fly with clamp+select (no divergent loads).
// =====================================================================
__device__ __forceinline__ float conv_gather(const float* __restrict__ X,
    int img, int oy, int ox, int kk, int Cin, int H, int W, int dec)
{
    int ci = kk >> 4;
    int t = kk & 15;
    int ky = t >> 2, kx = t & 3;
    bool valid = (ci < Cin);
    int iy, ix;
    if (!dec) {
        iy = oy * 2 + ky - 1;
        ix = ox * 2 + kx - 1;
    } else {
        int ty = oy + 1 - ky, tx = ox + 1 - kx;
        valid = valid && ty >= 0 && tx >= 0 && !(ty & 1) && !(tx & 1);
        iy = ty >> 1; ix = tx >> 1;
    }
    valid = valid && iy >= 0 && iy < H && ix >= 0 && ix < W;
    size_t idx = valid ? (((size_t)(img * H + iy) * W + ix) * Cin + ci) : 0;
    float v = X[idx];                 // unconditional load (idx clamped)
    return valid ? v : 0.0f;
}

__device__ __forceinline__ v16h conv_afrag(const float* __restrict__ X,
    int img, int oy, int ox, int k0, int half, int Cin, int H, int W, int dec)
{
    v16h f;
#pragma unroll
    for (int v = 0; v < 8; ++v) {
        int kb = ((v < 4) ? (2 * v) : (16 + 2 * (v - 4))) + 8 * half;
        f[2 * v]     = (_Float16)conv_gather(X, img, oy, ox, k0 + kb,     Cin, H, W, dec);
        f[2 * v + 1] = (_Float16)conv_gather(X, img, oy, ox, k0 + kb + 1, Cin, H, W, dec);
    }
    return f;
}

__global__ __launch_bounds__(32) void k_conv_gemm(
    const float* __restrict__ X, const _Float16* __restrict__ Bm,
    const float* __restrict__ bias, float* __restrict__ Cm,
    int NIMG, int Cin, int H, int W, int OH, int OW, int Cout, int Coutpad,
    int Kpad, int act, int dec)
{
    int M = NIMG * OH * OW;
    int N = Cout;
    int m0 = blockIdx.x * 32;
    int n0 = blockIdx.y * 32;
    int lane = threadIdx.x;
    int half = lane >> 4;
    int l16  = lane & 15;

    int ar0 = m0 + l16;       int ar0c = (ar0 < M) ? ar0 : (M - 1);
    int ar1 = m0 + 16 + l16;  int ar1c = (ar1 < M) ? ar1 : (M - 1);
    int img0 = ar0c / (OH * OW); int rr0 = ar0c - img0 * OH * OW;
    int oy0 = rr0 / OW, ox0 = rr0 - oy0 * OW;
    int img1 = ar1c / (OH * OW); int rr1 = ar1c - img1 * OH * OW;
    int oy1 = rr1 / OW, ox1 = rr1 - oy1 * OW;

    const _Float16* bp = Bm + (size_t)(8 * half) * Coutpad + n0 + l16;
    const size_t bstep = (size_t)32 * Coutpad;

    v8f acc00 = {}, acc01 = {}, acc10 = {}, acc11 = {};
    for (int k0 = 0; k0 < Kpad; k0 += 32) {
        v16h af0 = conv_afrag(X, img0, oy0, ox0, k0, half, Cin, H, W, dec);
        v16h af1 = conv_afrag(X, img1, oy1, ox1, k0, half, Cin, H, W, dec);
        v16h bf0, bf1;
#pragma unroll
        for (int s = 0; s < 16; ++s) { const _Float16* q = bp + (size_t)((s < 8) ? s : s + 8) * Coutpad; bf0[s] = q[0]; bf1[s] = q[16]; }
        __builtin_prefetch((const void*)(bp + bstep), 0, 1);
        acc00 = WMMA_F16(af0, bf0, acc00);
        acc01 = WMMA_F16(af0, bf1, acc01);
        acc10 = WMMA_F16(af1, bf0, acc10);
        acc11 = WMMA_F16(af1, bf1, acc11);
        bp += bstep;
    }

    int c0 = n0 + l16, c1 = n0 + 16 + l16;
    float bs0 = (bias && c0 < N) ? bias[c0] : 0.0f;
    float bs1 = (bias && c1 < N) ? bias[c1] : 0.0f;
#pragma unroll
    for (int j = 0; j < 8; ++j) {
        int r0 = m0 + j + 8 * half;
        int r1 = r0 + 16;
        if (r0 < M) {
            if (c0 < N) { float x = acc00[j] + bs0; if (act) x = siluf(x); Cm[(size_t)r0 * N + c0] = x; }
            if (c1 < N) { float x = acc01[j] + bs1; if (act) x = siluf(x); Cm[(size_t)r0 * N + c1] = x; }
        }
        if (r1 < M) {
            if (c0 < N) { float x = acc10[j] + bs0; if (act) x = siluf(x); Cm[(size_t)r1 * N + c0] = x; }
            if (c1 < N) { float x = acc11[j] + bs1; if (act) x = siluf(x); Cm[(size_t)r1 * N + c1] = x; }
        }
    }
}

// =====================================================================
// Weight prep kernels (f32 -> f16, layout transforms, K and N padding)
// =====================================================================
__global__ void k_prep_dense(const float* __restrict__ s, _Float16* __restrict__ d,
                             int K, int Kpad, int N, int Npad, long total)
{
    long i = (long)blockIdx.x * 256 + threadIdx.x;
    if (i >= total) return;
    int k = (int)(i / Npad), n = (int)(i - (long)k * Npad);
    d[i] = (k < K && n < N) ? (_Float16)s[(size_t)k * N + n] : (_Float16)0.0f;
}

// OIHW [Cout,Cin,4,4] -> [Kpad, Coutpad], k = ci*16 + ky*4 + kx
__global__ void k_prep_convw(const float* __restrict__ s, _Float16* __restrict__ d,
                             int Cin, int Cout, int Coutpad, int Kpad)
{
    long total = (long)Kpad * Coutpad;
    long i = (long)blockIdx.x * 256 + threadIdx.x;
    if (i >= total) return;
    int k = (int)(i / Coutpad), co = (int)(i - (long)k * Coutpad);
    int ci = k >> 4, t = k & 15;
    d[i] = (ci < Cin && co < Cout) ? (_Float16)s[(((size_t)co * Cin + ci) << 4) + t]
                                   : (_Float16)0.0f;
}

// enc_W rows permuted NCHW->NHWC: dst[(p*256+c)*1024+n] = src[(c*16+p)*1024+n]
__global__ void k_prep_encw(const float* __restrict__ s, _Float16* __restrict__ d)
{
    long total = 4096L * 1024;
    long i = (long)blockIdx.x * 256 + threadIdx.x;
    if (i >= total) return;
    int k = (int)(i / 1024), n = (int)(i % 1024);
    int p = k >> 8, c = k & 255;
    d[i] = (_Float16)s[(size_t)(c * 16 + p) * 1024 + n];
}

// dec_W columns permuted NCHW->NHWC: dst[k*4096 + p*256+c] = src[k*4096 + c*16+p]
__global__ void k_prep_decw(const float* __restrict__ s, _Float16* __restrict__ d)
{
    long total = 1536L * 4096;
    long i = (long)blockIdx.x * 256 + threadIdx.x;
    if (i >= total) return;
    int k = (int)(i / 4096), j = (int)(i % 4096);
    int p = j >> 8, c = j & 255;
    d[i] = (_Float16)s[(size_t)k * 4096 + c * 16 + p];
}
__global__ void k_perm_decb(const float* __restrict__ s, float* __restrict__ d)
{
    int i = blockIdx.x * 256 + threadIdx.x;
    if (i >= 4096) return;
    int p = i >> 8, c = i & 255;
    d[i] = s[c * 16 + p];
}

// =====================================================================
// Elementwise / small kernels
// =====================================================================
__global__ void k_preproc(const float* __restrict__ obs, float* __restrict__ x0)
{
    long total = 800L * 3 * 64 * 64;
    long i = (long)blockIdx.x * 256 + threadIdx.x;
    if (i >= total) return;
    long img = i / 12288; int r = (int)(i % 12288);
    int c = r / 4096, p = r % 4096;
    int y = p >> 6, x = p & 63;
    x0[((img * 64 + y) * 64 + x) * 3 + c] = obs[i] * (1.0f / 255.0f) - 0.5f;
}

__global__ void k_post_softmax(const float* __restrict__ lz, const float* __restrict__ temp,
                               float* __restrict__ stoch)
{
    int g = blockIdx.x * 256 + threadIdx.x;
    if (g >= 800 * 32) return;
    float inv_t = 1.0f / temp[0];
    const float* p = lz + (size_t)g * 32;
    float mx = -1e30f;
    for (int j = 0; j < 32; ++j) mx = fmaxf(mx, p[j] * inv_t);
    float s = 0.0f;
    for (int j = 0; j < 32; ++j) s += __expf(p[j] * inv_t - mx);
    float inv_s = 1.0f / s;
    float* o = stoch + (size_t)g * 32;
    for (int j = 0; j < 32; ++j)
        o[j] = 0.99f * __expf(p[j] * inv_t - mx) * inv_s + 0.01f / 32.0f;
}

__global__ void k_build_tok(const float* __restrict__ stoch, const float* __restrict__ gt,
                            float* __restrict__ tok)
{
    long total = 784L * 1056;
    long i = (long)blockIdx.x * 256 + threadIdx.x;
    if (i >= total) return;
    int row = (int)(i / 1056), j = (int)(i % 1056);
    int b = row / SSq, t = row % SSq;
    float v = 0.0f;
    if (j < 1024)       v = stoch[(size_t)(b * TT + t) * SSD + j];
    else if (j < 1030)  v = gt[(size_t)(b * TT + t) * AA + (j - 1024)];
    tok[i] = v;
}

__global__ void k_add_pos(float* __restrict__ h, const float* __restrict__ pos)
{
    long total = 784L * DM;
    long i = (long)blockIdx.x * 256 + threadIdx.x;
    if (i >= total) return;
    int row = (int)(i / DM), n = (int)(i % DM);
    int t = row % SSq;
    h[i] += pos[(size_t)t * DM + n];
}

__global__ __launch_bounds__(256) void k_rms(const float* __restrict__ x,
    const float* __restrict__ g, float* __restrict__ y, int D, int dosilu)
{
    int row = blockIdx.x;
    const float* xr = x + (size_t)row * D;
    __shared__ float red[256];
    float s = 0.0f;
    for (int i = threadIdx.x; i < D; i += 256) { float v = xr[i]; s += v * v; }
    red[threadIdx.x] = s;
    __syncthreads();
    for (int st = 128; st > 0; st >>= 1) {
        if (threadIdx.x < st) red[threadIdx.x] += red[threadIdx.x + st];
        __syncthreads();
    }
    float rs = rsqrtf(red[0] / (float)D + 1e-6f);
    for (int i = threadIdx.x; i < D; i += 256) {
        float v = xr[i] * rs * g[i];
        if (dosilu) v = siluf(v);
        y[(size_t)row * D + i] = v;
    }
}

__global__ __launch_bounds__(64) void k_attn(const float* __restrict__ qkv, float* __restrict__ o)
{
    int bh = blockIdx.x;
    int b = bh >> 3, hh = bh & 7;
    int qi = blockIdx.y;
    int t = threadIdx.x;
    __shared__ float sc[64];
    const float* qv = qkv + ((size_t)(b * SSq + qi)) * 1536 + hh * 64;
    float s = -1e30f;
    if (t < SSq && t <= qi) {
        const float* kv = qkv + ((size_t)(b * SSq + t)) * 1536 + 512 + hh * 64;
        float d = 0.0f;
        for (int i = 0; i < 64; ++i) d += qv[i] * kv[i];
        s = d * 0.125f;
    }
    sc[t] = s;
    __syncthreads();
    float mx = -1e30f;
    for (int k = 0; k <= qi; ++k) mx = fmaxf(mx, sc[k]);
    float sum = 0.0f;
    for (int k = 0; k <= qi; ++k) sum += __expf(sc[k] - mx);
    float inv = 1.0f / sum;
    float acc = 0.0f;
    for (int k = 0; k <= qi; ++k) {
        float p = __expf(sc[k] - mx) * inv;
        acc += p * qkv[((size_t)(b * SSq + k)) * 1536 + 1024 + hh * 64 + t];
    }
    o[((size_t)((b * SSq + qi) * NH + hh)) * 64 + t] = acc;
}

__global__ void k_concat_feat(const float* __restrict__ h, const float* __restrict__ stoch,
                              float* __restrict__ feat)
{
    long total = 784L * DENSE_IN;
    long i = (long)blockIdx.x * 256 + threadIdx.x;
    if (i >= total) return;
    int row = (int)(i / DENSE_IN), j = (int)(i % DENSE_IN);
    int b = row / SSq, t = row % SSq;
    feat[i] = (j < DM) ? h[(size_t)row * DM + j]
                       : stoch[(size_t)(b * TT + t + 1) * SSD + (j - DM)];
}

__global__ void k_actfin(const float* __restrict__ pl, float* __restrict__ logits,
                         float* __restrict__ oh)
{
    int row = blockIdx.x * 256 + threadIdx.x;
    if (row >= 784) return;
    const float* p = pl + (size_t)row * AA;
    float mx = -1e30f;
    for (int j = 0; j < AA; ++j) mx = fmaxf(mx, p[j]);
    float e[AA]; float s = 0.0f;
    for (int j = 0; j < AA; ++j) { e[j] = __expf(p[j] - mx); s += e[j]; }
    float inv = 1.0f / s;
    float best = -1.0f; int bi = 0;
    for (int j = 0; j < AA; ++j) {
        float pr = 0.99f * e[j] * inv + 0.01f / (float)AA;
        logits[(size_t)row * AA + j] = __logf(pr);
        if (pr > best) { best = pr; bi = j; }
    }
    for (int j = 0; j < AA; ++j) oh[(size_t)row * AA + j] = (j == bi) ? 1.0f : 0.0f;
}

__global__ void k_pred_action(const float* __restrict__ oh, float* __restrict__ out)
{
    int i = blockIdx.x * 256 + threadIdx.x;
    if (i >= BB * TT * AA) return;
    int b = i / (TT * AA);
    int r = i % (TT * AA);
    int t = r / AA, a = r % AA;
    float v = 0.0f;
    if (t >= 1 && t <= SSq - 1) v = oh[(size_t)(b * SSq + (t - 1)) * AA + a];
    out[i] = v;
}

__global__ void k_copy(const float* __restrict__ s, float* __restrict__ d, long n)
{
    long i = (long)blockIdx.x * 256 + threadIdx.x;
    if (i < n) d[i] = s[i];
}

__global__ void k_recon_perm(const float* __restrict__ src, float* __restrict__ dst)
{
    long total = 784L * 3 * 4096;
    long i = (long)blockIdx.x * 256 + threadIdx.x;
    if (i >= total) return;
    long bs = i / 12288; int r = (int)(i % 12288);
    int c = r / 4096, p = r % 4096;
    int y = p >> 6, x = p & 63;
    dst[i] = src[((bs * 64 + y) * 64 + x) * 3 + c];
}

// =====================================================================
// Host orchestration
// =====================================================================
static inline size_t al256(size_t x) { return (x + 255) & ~(size_t)255; }
static inline dim3 g1(long n) { return dim3((unsigned)((n + 255) / 256)); }
static inline dim3 gemm_grid(int M, int N) { return dim3((unsigned)((M + 31) / 32), (unsigned)((N + 31) / 32)); }

extern "C" void kernel_launch(void* const* d_in, const int* in_sizes, int n_in,
                              void* d_out, int out_size, void* d_ws, size_t ws_size,
                              hipStream_t stream)
{
    (void)in_sizes; (void)n_in; (void)out_size; (void)ws_size;
    const float* obs     = (const float*)d_in[0];
    const float* temp    = (const float*)d_in[2];
    const float* gt      = (const float*)d_in[3];
    const float* ck1 = (const float*)d_in[4];  const float* cb1 = (const float*)d_in[5];
    const float* ck2 = (const float*)d_in[6];  const float* cb2 = (const float*)d_in[7];
    const float* ck3 = (const float*)d_in[8];  const float* cb3 = (const float*)d_in[9];
    const float* ck4 = (const float*)d_in[10]; const float* cb4 = (const float*)d_in[11];
    const float* enc_W = (const float*)d_in[12]; const float* enc_b = (const float*)d_in[13];
    const float* post_W = (const float*)d_in[14]; const float* post_b = (const float*)d_in[15];
    const float* in_W = (const float*)d_in[16]; const float* in_b = (const float*)d_in[17];
    const float* pos_emb = (const float*)d_in[18];
    const float* ln1 = (const float*)d_in[19]; const float* ln2 = (const float*)d_in[20];
    const float* qkv_W = (const float*)d_in[21]; const float* qkv_b = (const float*)d_in[22];
    const float* o_W = (const float*)d_in[23]; const float* o_b = (const float*)d_in[24];
    const float* ff1_W = (const float*)d_in[25]; const float* ff1_b = (const float*)d_in[26];
    const float* ff2_W = (const float*)d_in[27]; const float* ff2_b = (const float*)d_in[28];
    const float* a0_W = (const float*)d_in[29]; const float* a0_b = (const float*)d_in[30];
    const float* a0_g = (const float*)d_in[31];
    const float* aWp = (const float*)d_in[32]; const float* ab = (const float*)d_in[33];
    const float* ag = (const float*)d_in[34];
    const float* aout_W = (const float*)d_in[35]; const float* aout_b = (const float*)d_in[36];
    const float* dec_W = (const float*)d_in[37]; const float* dec_b = (const float*)d_in[38];
    const float* dk1 = (const float*)d_in[39]; const float* db1 = (const float*)d_in[40];
    const float* dk2 = (const float*)d_in[41]; const float* db2 = (const float*)d_in[42];
    const float* dk3 = (const float*)d_in[43]; const float* db3 = (const float*)d_in[44];
    const float* dk4 = (const float*)d_in[45]; const float* db4 = (const float*)d_in[46];

    float* out = (float*)d_out;
    const long AL_OFF  = 0;
    const long REC_OFF = 4704;
    const long DET_OFF = REC_OFF + 9633792L;
    const long PRED_OFF = DET_OFF + 401408L;

    // ---- workspace bump allocator ----
    char* wsb = (char*)d_ws;
    size_t off = 0;
    auto wsalloc = [&](size_t bytes) -> void* { void* p = wsb + off; off = al256(off + bytes); return p; };

    float* arenaP = (float*)wsalloc(13107200UL * 4);   // big ping
    float* arenaQ = (float*)wsalloc(26214400UL * 4);   // big pong
    float* emb    = (float*)wsalloc(819200UL * 4);
    float* lzb    = (float*)wsalloc(819200UL * 4);
    float* stoch  = (float*)wsalloc(819200UL * 4);
    float* tok    = (float*)wsalloc(827904UL * 4);
    float* hbuf   = (float*)wsalloc(401408UL * 4);
    float* ybuf   = (float*)wsalloc(401408UL * 4);
    float* qkvb   = (float*)wsalloc(1204224UL * 4);
    float* obuf   = (float*)wsalloc(401408UL * 4);
    float* f1b    = (float*)wsalloc(1605632UL * 4);
    float* feat   = (float*)wsalloc(1204224UL * 4);
    float* acur   = (float*)wsalloc(802816UL * 4);
    float* atmp   = (float*)wsalloc(802816UL * 4);
    float* plb    = (float*)wsalloc(4704UL * 4);
    float* ohb    = (float*)wsalloc(4704UL * 4);
    float* decbp  = (float*)wsalloc(4096UL * 4);
    _Float16* wc1  = (_Float16*)wsalloc(64UL * 32 * 2);
    _Float16* wc2  = (_Float16*)wsalloc(512UL * 64 * 2);
    _Float16* wc3  = (_Float16*)wsalloc(1024UL * 128 * 2);
    _Float16* wc4  = (_Float16*)wsalloc(2048UL * 256 * 2);
    _Float16* wenc = (_Float16*)wsalloc(4096UL * 1024 * 2);
    _Float16* wpost= (_Float16*)wsalloc(1024UL * 1024 * 2);
    _Float16* win  = (_Float16*)wsalloc(1056UL * 512 * 2);
    _Float16* wqkv = (_Float16*)wsalloc(2UL * 512 * 1536 * 2);
    _Float16* wo   = (_Float16*)wsalloc(2UL * 512 * 512 * 2);
    _Float16* wff1 = (_Float16*)wsalloc(2UL * 512 * 2048 * 2);
    _Float16* wff2 = (_Float16*)wsalloc(2UL * 2048 * 512 * 2);
    _Float16* wa0  = (_Float16*)wsalloc(1536UL * 1024 * 2);
    _Float16* waW  = (_Float16*)wsalloc(4UL * 1024 * 1024 * 2);
    _Float16* waout= (_Float16*)wsalloc(1024UL * 32 * 2);
    _Float16* wdec = (_Float16*)wsalloc(1536UL * 4096 * 2);
    _Float16* wd1  = (_Float16*)wsalloc(4096UL * 128 * 2);
    _Float16* wd2  = (_Float16*)wsalloc(2048UL * 64 * 2);
    _Float16* wd3  = (_Float16*)wsalloc(1024UL * 32 * 2);
    _Float16* wd4  = (_Float16*)wsalloc(512UL * 32 * 2);

    // ---- weight prep ----
    k_prep_convw<<<g1(64L * 32), 256, 0, stream>>>(ck1, wc1, 3, 32, 32, 64);
    k_prep_convw<<<g1(512L * 64), 256, 0, stream>>>(ck2, wc2, 32, 64, 64, 512);
    k_prep_convw<<<g1(1024L * 128), 256, 0, stream>>>(ck3, wc3, 64, 128, 128, 1024);
    k_prep_convw<<<g1(2048L * 256), 256, 0, stream>>>(ck4, wc4, 128, 256, 256, 2048);
    k_prep_encw<<<g1(4096L * 1024), 256, 0, stream>>>(enc_W, wenc);
    k_prep_dense<<<g1(1024L * 1024), 256, 0, stream>>>(post_W, wpost, 1024, 1024, 1024, 1024, 1024L * 1024);
    k_prep_dense<<<g1(1056L * 512), 256, 0, stream>>>(in_W, win, 1030, 1056, 512, 512, 1056L * 512);
    k_prep_dense<<<g1(1024L * 1536), 256, 0, stream>>>(qkv_W, wqkv, 1024, 1024, 1536, 1536, 1024L * 1536);
    k_prep_dense<<<g1(1024L * 512), 256, 0, stream>>>(o_W, wo, 1024, 1024, 512, 512, 1024L * 512);
    k_prep_dense<<<g1(1024L * 2048), 256, 0, stream>>>(ff1_W, wff1, 1024, 1024, 2048, 2048, 1024L * 2048);
    k_prep_dense<<<g1(4096L * 512), 256, 0, stream>>>(ff2_W, wff2, 4096, 4096, 512, 512, 4096L * 512);
    k_prep_dense<<<g1(1536L * 1024), 256, 0, stream>>>(a0_W, wa0, 1536, 1536, 1024, 1024, 1536L * 1024);
    k_prep_dense<<<g1(4096L * 1024), 256, 0, stream>>>(aWp, waW, 4096, 4096, 1024, 1024, 4096L * 1024);
    k_prep_dense<<<g1(1024L * 32), 256, 0, stream>>>(aout_W, waout, 1024, 1024, 6, 32, 1024L * 32);
    k_prep_decw<<<g1(1536L * 4096), 256, 0, stream>>>(dec_W, wdec);
    k_perm_decb<<<g1(4096), 256, 0, stream>>>(dec_b, decbp);
    k_prep_convw<<<g1(4096L * 128), 256, 0, stream>>>(dk1, wd1, 256, 128, 128, 4096);
    k_prep_convw<<<g1(2048L * 64), 256, 0, stream>>>(dk2, wd2, 128, 64, 64, 2048);
    k_prep_convw<<<g1(1024L * 32), 256, 0, stream>>>(dk3, wd3, 64, 32, 32, 1024);
    k_prep_convw<<<g1(512L * 32), 256, 0, stream>>>(dk4, wd4, 32, 3, 32, 512);

    // ---- encoder ----
    float* x0 = arenaP;  // [800,64,64,3]
    k_preproc<<<g1(800L * 3 * 64 * 64), 256, 0, stream>>>(obs, x0);
    float* y1 = arenaQ;  // [800,32,32,32]
    k_conv_gemm<<<gemm_grid(800 * 1024, 32), 32, 0, stream>>>(x0, wc1, cb1, y1, 800, 3, 64, 64, 32, 32, 32, 32, 64, 1, 0);
    float* y2 = arenaP;  // [800,16,16,64]
    k_conv_gemm<<<gemm_grid(800 * 256, 64), 32, 0, stream>>>(y1, wc2, cb2, y2, 800, 32, 32, 32, 16, 16, 64, 64, 512, 1, 0);
    float* y3 = arenaQ;  // [800,8,8,128]
    k_conv_gemm<<<gemm_grid(800 * 64, 128), 32, 0, stream>>>(y2, wc3, cb3, y3, 800, 64, 16, 16, 8, 8, 128, 128, 1024, 1, 0);
    float* y4 = arenaP;  // [800,4,4,256]
    k_conv_gemm<<<gemm_grid(800 * 16, 256), 32, 0, stream>>>(y3, wc4, cb4, y4, 800, 128, 8, 8, 4, 4, 256, 256, 2048, 1, 0);
    k_gemm<<<gemm_grid(800, 1024), 32, 0, stream>>>(y4, wenc, enc_b, nullptr, emb, 800, 1024, 1024, 4096, 1);
    k_gemm<<<gemm_grid(800, 1024), 32, 0, stream>>>(emb, wpost, post_b, nullptr, lzb, 800, 1024, 1024, 1024, 0);
    k_post_softmax<<<g1(800L * 32), 256, 0, stream>>>(lzb, temp, stoch);

    // ---- transformer prior ----
    k_build_tok<<<g1(784L * 1056), 256, 0, stream>>>(stoch, gt, tok);
    k_gemm<<<gemm_grid(784, 512), 32, 0, stream>>>(tok, win, in_b, nullptr, hbuf, 784, 512, 512, 1056, 0);
    k_add_pos<<<g1(784L * 512), 256, 0, stream>>>(hbuf, pos_emb);
    for (int l = 0; l < 2; ++l) {
        k_rms<<<784, 256, 0, stream>>>(hbuf, ln1 + (size_t)l * 512, ybuf, 512, 0);
        k_gemm<<<gemm_grid(784, 1536), 32, 0, stream>>>(ybuf, wqkv + (size_t)l * 512 * 1536,
                                                        qkv_b + (size_t)l * 1536, nullptr, qkvb, 784, 1536, 1536, 512, 0);
        k_attn<<<dim3(BB * NH, SSq), 64, 0, stream>>>(qkvb, obuf);
        k_gemm<<<gemm_grid(784, 512), 32, 0, stream>>>(obuf, wo + (size_t)l * 512 * 512,
                                                       o_b + (size_t)l * 512, hbuf, hbuf, 784, 512, 512, 512, 0);
        k_rms<<<784, 256, 0, stream>>>(hbuf, ln2 + (size_t)l * 512, ybuf, 512, 0);
        k_gemm<<<gemm_grid(784, 2048), 32, 0, stream>>>(ybuf, wff1 + (size_t)l * 512 * 2048,
                                                        ff1_b + (size_t)l * 2048, nullptr, f1b, 784, 2048, 2048, 512, 1);
        k_gemm<<<gemm_grid(784, 512), 32, 0, stream>>>(f1b, wff2 + (size_t)l * 2048 * 512,
                                                       ff2_b + (size_t)l * 512, hbuf, hbuf, 784, 512, 512, 2048, 0);
    }
    k_copy<<<g1(401408L), 256, 0, stream>>>(hbuf, out + DET_OFF, 401408L);

    // ---- feature + action MLP ----
    k_concat_feat<<<g1(784L * 1536), 256, 0, stream>>>(hbuf, stoch, feat);
    k_gemm<<<gemm_grid(784, 1024), 32, 0, stream>>>(feat, wa0, a0_b, nullptr, atmp, 784, 1024, 1024, 1536, 0);
    k_rms<<<784, 256, 0, stream>>>(atmp, a0_g, acur, 1024, 1);
    for (int i = 0; i < 4; ++i) {
        k_gemm<<<gemm_grid(784, 1024), 32, 0, stream>>>(acur, waW + (size_t)i * 1024 * 1024,
                                                        ab + (size_t)i * 1024, nullptr, atmp, 784, 1024, 1024, 1024, 0);
        k_rms<<<784, 256, 0, stream>>>(atmp, ag + (size_t)i * 1024, acur, 1024, 1);
    }
    k_gemm<<<gemm_grid(784, 6), 32, 0, stream>>>(acur, waout, aout_b, nullptr, plb, 784, 6, 32, 1024, 0);
    k_actfin<<<g1(784), 256, 0, stream>>>(plb, out + AL_OFF, ohb);
    k_pred_action<<<g1(BB * TT * AA), 256, 0, stream>>>(ohb, out + PRED_OFF);

    // ---- conv decoder ----
    float* d0 = arenaP;  // [784,4,4,256] NHWC
    k_gemm<<<gemm_grid(784, 4096), 32, 0, stream>>>(feat, wdec, decbp, nullptr, d0, 784, 4096, 4096, 1536, 0);
    float* dd1 = arenaQ; // [784,8,8,128]
    k_conv_gemm<<<gemm_grid(784 * 64, 128), 32, 0, stream>>>(d0, wd1, db1, dd1, 784, 256, 4, 4, 8, 8, 128, 128, 4096, 1, 1);
    float* dd2 = arenaP; // [784,16,16,64]
    k_conv_gemm<<<gemm_grid(784 * 256, 64), 32, 0, stream>>>(dd1, wd2, db2, dd2, 784, 128, 8, 8, 16, 16, 64, 64, 2048, 1, 1);
    float* dd3 = arenaQ; // [784,32,32,32]
    k_conv_gemm<<<gemm_grid(784 * 1024, 32), 32, 0, stream>>>(dd2, wd3, db3, dd3, 784, 64, 16, 16, 32, 32, 32, 32, 1024, 1, 1);
    float* dd4 = arenaP; // [784,64,64,3]
    k_conv_gemm<<<gemm_grid(784 * 4096, 3), 32, 0, stream>>>(dd3, wd4, db4, dd4, 784, 32, 32, 32, 64, 64, 3, 32, 512, 0, 1);
    k_recon_perm<<<g1(784L * 3 * 4096), 256, 0, stream>>>(dd4, out + REC_OFF);
}
